// WrapperYukarinSa_26542897889539
// MI455X (gfx1250) — matrix operand order repaired and checked by reading, not compiled, not run on hardware
//
#include <hip/hip_runtime.h>
#include <hip/hip_bf16.h>
#include <stdint.h>

// ---------------- problem constants ----------------
constexpr int kB    = 256;
constexpr int kT    = 1024;
constexpr int kE    = 64;
constexpr int kSpk  = 64;
constexpr int kC    = 256;      // encoder channels
constexpr int kInCh = 132;
constexpr int kH    = 256;      // GRU hidden
constexpr int kG    = 768;      // 3*H
constexpr int kKConv = 396;     // 132*3
constexpr int kKPad  = 416;     // padded to mult of 32
constexpr long long kM = (long long)kB * kT;   // 262144 GEMM rows
constexpr int kNWG  = 16;       // sequential-phase workgroups (cluster-sized)

typedef unsigned short ushort_t;
typedef __attribute__((ext_vector_type(16))) __bf16 v16bf;
typedef __attribute__((ext_vector_type(8)))  float  v8f;

union Frag { v16bf v; unsigned int u[8]; };

// ---------------- small helpers ----------------
__device__ __forceinline__ ushort_t f2bf(float x) {
  unsigned int u = __float_as_uint(x);
  unsigned int r = (u + 0x7FFFu + ((u >> 16) & 1u)) >> 16;   // RNE
  return (ushort_t)r;
}
__device__ __forceinline__ float bf2f(ushort_t h) {
  return __uint_as_float(((unsigned int)h) << 16);
}

__device__ __forceinline__ v8f wmma_bf16(v16bf a, v16bf b, v8f c) {
  return __builtin_amdgcn_wmma_f32_16x16x32_bf16(false, a, false, b, (short)0, c, false, false);
}

// A fragment: 16x32 bf16 tile, rows m0..m0+15, cols k0..k0+31, row-major src (ldK elems).
// Per ISA layout: lanes 0-15 -> K chunks {0..7,16..23}; lanes 16-31 -> {8..15,24..31}.
__device__ __forceinline__ v16bf load_a_frag_u128(const ushort_t* __restrict__ A, int ldK,
                                                  int m0, int k0) {
  int l = threadIdx.x & 31;
  const ushort_t* p = A + (size_t)(m0 + (l & 15)) * ldK + k0 + ((l < 16) ? 0 : 8);
  Frag f;
  *(uint4*)&f.u[0] = *(const uint4*)p;
  *(uint4*)&f.u[4] = *(const uint4*)(p + 16);
  return f.v;
}

// A fragment from a per-lane row pointer (row already selected), 4-byte-aligned source.
__device__ __forceinline__ v16bf load_a_frag_row(const ushort_t* __restrict__ arow, int k0) {
  int l = threadIdx.x & 31;
  const ushort_t* p = arow + k0 + ((l < 16) ? 0 : 8);
  Frag f;
#pragma unroll
  for (int i = 0; i < 4; ++i) f.u[i] = *(const unsigned int*)(p + 2 * i);
#pragma unroll
  for (int i = 0; i < 4; ++i) f.u[4 + i] = *(const unsigned int*)(p + 16 + 2 * i);
  return f.v;
}

// B fragment: 32x16 bf16 tile (KxN); source stored N-major: row n holds contiguous K (ldK elems).
// lanes 0-15: K={0..7,16..23} for col n=lane; lanes 16-31: K={8..15,24..31} for col n=lane-16.
__device__ __forceinline__ v16bf load_b_frag(const ushort_t* __restrict__ Bm, int ldK,
                                             int n0, int k0) {
  int l = threadIdx.x & 31;
  const ushort_t* p = Bm + (size_t)(n0 + (l & 15)) * ldK + k0 + ((l < 16) ? 0 : 8);
  Frag f;
  *(uint4*)&f.u[0] = *(const uint4*)p;
  *(uint4*)&f.u[4] = *(const uint4*)(p + 16);
  return f.v;
}

// ---------------- CDNA5 feature guards ----------------
#if __has_builtin(__builtin_amdgcn_tensor_load_to_lds)
#define HAVE_TDM 1
typedef __attribute__((ext_vector_type(4))) unsigned int v4u;
typedef __attribute__((ext_vector_type(8))) int v8i;
typedef __attribute__((ext_vector_type(4))) int v4i;
#endif

__device__ __forceinline__ void backoff() {
#if __has_builtin(__builtin_amdgcn_s_sleep)
  __builtin_amdgcn_s_sleep(1);
#endif
}
__device__ __forceinline__ void cluster_barrier_hint() {
#if __has_builtin(__builtin_amdgcn_s_cluster_barrier)
  __builtin_amdgcn_s_cluster_barrier();   // hardware fast path; NOP if not clustered
#endif
}
__device__ __forceinline__ void wait_tensorcnt0() {
#if __has_builtin(__builtin_amdgcn_s_wait_tensorcnt)
  __builtin_amdgcn_s_wait_tensorcnt(0);
#endif
}

// fenced grid barrier across kNWG workgroups (monotonic counter, reusable)
__device__ __forceinline__ void grid_barrier(unsigned int* cnt) {
  __threadfence();          // release our stores to device scope
  __syncthreads();
  if (threadIdx.x == 0) {
    cluster_barrier_hint();
    unsigned arrival = __hip_atomic_fetch_add(cnt, 1u, __ATOMIC_ACQ_REL,
                                              __HIP_MEMORY_SCOPE_AGENT);
    unsigned target = (arrival / kNWG + 1u) * (unsigned)kNWG;
    while (__hip_atomic_load(cnt, __ATOMIC_ACQUIRE, __HIP_MEMORY_SCOPE_AGENT) < target)
      backoff();
  }
  __syncthreads();
  __threadfence();          // acquire: invalidate L0 before reading others' data
}

// ---------------- K0a: pack weights to bf16, init state ----------------
__global__ void k_prep(const float* __restrict__ enc_w, const float* __restrict__ W_ih,
                       const float* __restrict__ W_hh, const float* __restrict__ hid0,
                       ushort_t* __restrict__ Bc, ushort_t* __restrict__ Wi,
                       ushort_t* __restrict__ Whh, float* __restrict__ wf0,
                       float* __restrict__ hidf, ushort_t* __restrict__ hidb,
                       float* __restrict__ f0, unsigned int* __restrict__ cnt) {
  int stride = gridDim.x * blockDim.x;
  int tid = blockIdx.x * blockDim.x + threadIdx.x;
  // conv weights: Bc[c][k*132+i] = enc_w[c][i][k], K padded with zeros to 416
  for (int idx = tid; idx < kC * kKPad; idx += stride) {
    int c = idx / kKPad, kk = idx % kKPad;
    float v = 0.f;
    if (kk < kKConv) { int k = kk / kInCh, i = kk % kInCh; v = enc_w[(c * kInCh + i) * 3 + k]; }
    Bc[idx] = f2bf(v);
  }
  // W_ih split: Wi[j][c]=W_ih[j][c] (c<256), wf0[j]=W_ih[j][256]; W_hh bf16 copy
  for (int idx = tid; idx < kG * kH; idx += stride) {
    int j = idx / kH, c = idx % kH;
    Wi[idx]  = f2bf(W_ih[j * (kH + 1) + c]);
    Whh[idx] = f2bf(W_hh[idx]);
  }
  for (int j = tid; j < kG; j += stride) wf0[j] = W_ih[j * (kH + 1) + kH];
  for (int idx = tid; idx < kB * kH; idx += stride) {
    float v = hid0[idx]; hidf[idx] = v; hidb[idx] = f2bf(v);
  }
  for (int idx = tid; idx < kB; idx += stride) f0[idx] = 0.f;
  if (tid == 0) *cnt = 0u;
}

// ---------------- K0b: materialize padded encoder input (B, T+2, 132) bf16 ----------------
__global__ void k_encin(const int* __restrict__ vlist, const int* __restrict__ clist,
                        const int* __restrict__ sa, const int* __restrict__ ea,
                        const int* __restrict__ sap, const int* __restrict__ eap,
                        const int* __restrict__ spk, const float* __restrict__ pht,
                        const float* __restrict__ spt, ushort_t* __restrict__ enc) {
  const long long N = (long long)kB * (kT + 2) * kInCh;
  const long long NTOT = N + 64;                      // tail pad for last-row overread
  long long stride = (long long)gridDim.x * blockDim.x;
  for (long long idx = blockIdx.x * (long long)blockDim.x + threadIdx.x;
       idx < NTOT; idx += stride) {
    float val = 0.f;
    if (idx < N) {
      int ch = (int)(idx % kInCh);
      long long row = idx / kInCh;
      int tl = (int)(row % (kT + 2));
      int b  = (int)(row / (kT + 2));
      if (tl >= 1 && tl <= kT) {
        long long bt = (long long)b * kT + (tl - 1);
        if (ch < kE) {
          val = pht[(vlist[bt] + 1) * kE + ch] + pht[(clist[bt] + 1) * kE + ch];
        } else if (ch < kE + 4) {
          int a = ch - kE;
          int f = (a == 0) ? sa[bt] : (a == 1) ? ea[bt] : (a == 2) ? sap[bt] : eap[bt];
          val = (float)f;
        } else {
          val = spt[spk[b] * kSpk + (ch - (kE + 4))];
        }
      }
    }
    enc[idx] = f2bf(val);
  }
}

// ---------------- K1: conv-as-GEMM + bias + ReLU -> h bf16 (M x 256) ----------------
// 2x2 register blocking: each wave computes a 32x32 output block.
__global__ void __launch_bounds__(256)
k_conv(const ushort_t* __restrict__ enc, const ushort_t* __restrict__ Bc,
       const float* __restrict__ enc_b, ushort_t* __restrict__ hOut) {
  unsigned wid = (blockIdx.x * blockDim.x + threadIdx.x) >> 5;
  const unsigned NB = kC / 32;                 // 8 n-blocks
  unsigned nb = wid % NB, mb = wid / NB;       // mb < 8192
  int l = threadIdx.x & 31;
  int mrow0 = (int)mb * 32 + (l & 15);
  int mrow1 = mrow0 + 16;
  const ushort_t* arow0 = enc + (size_t)((mrow0 >> 10) * (kT + 2) + (mrow0 & (kT - 1))) * kInCh;
  const ushort_t* arow1 = enc + (size_t)((mrow1 >> 10) * (kT + 2) + (mrow1 & (kT - 1))) * kInCh;
  v8f c00 = {0.f,0.f,0.f,0.f,0.f,0.f,0.f,0.f};
  v8f c01 = c00, c10 = c00, c11 = c00;
#pragma unroll 1
  for (int kk = 0; kk < kKPad / 32; ++kk) {
    int k0 = kk * 32;
    v16bf a0 = load_a_frag_row(arow0, k0);
    v16bf a1 = load_a_frag_row(arow1, k0);
    v16bf b0 = load_b_frag(Bc, kKPad, (int)nb * 32, k0);
    v16bf b1 = load_b_frag(Bc, kKPad, (int)nb * 32 + 16, k0);
    c00 = wmma_bf16(a0, b0, c00);
    c01 = wmma_bf16(a0, b1, c01);
    c10 = wmma_bf16(a1, b0, c10);
    c11 = wmma_bf16(a1, b1, c11);
  }
  v8f acc[2][2] = {{c00, c01}, {c10, c11}};
#pragma unroll
  for (int mi = 0; mi < 2; ++mi) {
#pragma unroll
    for (int ni = 0; ni < 2; ++ni) {
      int n = (int)nb * 32 + ni * 16 + (l & 15);
      float bias = enc_b[n];
      int mbase = (int)mb * 32 + mi * 16 + ((l < 16) ? 0 : 8);
#pragma unroll
      for (int r = 0; r < 8; ++r) {
        float x = acc[mi][ni][r] + bias;
        x = x > 0.f ? x : 0.f;
        hOut[(size_t)(mbase + r) * kC + n] = f2bf(x);
      }
    }
  }
}

// ---------------- K2: gi = h @ Wi^T + b_ih -> bf16 (M x 768) ----------------
// 2x2 register blocking: each wave computes a 32x32 output block.
__global__ void __launch_bounds__(256)
k_gi(const ushort_t* __restrict__ h, const ushort_t* __restrict__ Wi,
     const float* __restrict__ b_ih, ushort_t* __restrict__ gi) {
  unsigned wid = (blockIdx.x * blockDim.x + threadIdx.x) >> 5;
  const unsigned NB = kG / 32;                 // 24 n-blocks
  unsigned nb = wid % NB, mb = wid / NB;       // mb < 8192
  int l = threadIdx.x & 31;
  v8f c00 = {0.f,0.f,0.f,0.f,0.f,0.f,0.f,0.f};
  v8f c01 = c00, c10 = c00, c11 = c00;
#pragma unroll
  for (int kk = 0; kk < kH / 32; ++kk) {
    int k0 = kk * 32;
    v16bf a0 = load_a_frag_u128(h, kC, (int)mb * 32, k0);
    v16bf a1 = load_a_frag_u128(h, kC, (int)mb * 32 + 16, k0);
    v16bf b0 = load_b_frag(Wi, kH, (int)nb * 32, k0);
    v16bf b1 = load_b_frag(Wi, kH, (int)nb * 32 + 16, k0);
    c00 = wmma_bf16(a0, b0, c00);
    c01 = wmma_bf16(a0, b1, c01);
    c10 = wmma_bf16(a1, b0, c10);
    c11 = wmma_bf16(a1, b1, c11);
  }
  v8f acc[2][2] = {{c00, c01}, {c10, c11}};
#pragma unroll
  for (int mi = 0; mi < 2; ++mi) {
#pragma unroll
    for (int ni = 0; ni < 2; ++ni) {
      int n = (int)nb * 32 + ni * 16 + (l & 15);
      float bias = b_ih[n];
      int mbase = (int)mb * 32 + mi * 16 + ((l < 16) ? 0 : 8);
#pragma unroll
      for (int r = 0; r < 8; ++r)
        gi[(size_t)(mbase + r) * kG + n] = f2bf(acc[mi][ni][r] + bias);
    }
  }
}

// ---------------- K3: sequential GRU over T steps, 16 cooperating WGs ----------------
__global__ void __launch_bounds__(256)
k_seq(const ushort_t* __restrict__ gi,     // [M][768] bf16 (b_ih folded in)
      const ushort_t* __restrict__ Whh,    // [768][256] bf16
      const float* __restrict__ wf0g,      // [768] f0 column of W_ih
      const float* __restrict__ b_hh,      // [768]
      const float* __restrict__ post_w,    // [256]
      const float* __restrict__ post_b,    // [1]
      float* __restrict__ hidf,            // [256][256] f32 hidden
      ushort_t* __restrict__ hidb,         // [256][256] bf16 hidden (matmul operand)
      float* __restrict__ gh,              // [256][768] per-step scratch
      float* __restrict__ f0,              // [256]
      unsigned int* __restrict__ cnt,
      float* __restrict__ out) {           // [B][T]
  __shared__ ushort_t sWhh[48 * kH];       // this WG's 48 rows of W_hh (24 KB)
  __shared__ float sWf0[kG];
  __shared__ float sBhh[kG];
  __shared__ float sPw[kH];
  __shared__ float sRed[256];
#ifdef HAVE_TDM
  __shared__ ushort_t sGi[16 * kG];        // TDM-prefetched gi tile (24 KB)
#endif
  const int g = blockIdx.x;
  const int tid = threadIdx.x;
  const int wave = tid >> 5;

  {  // preload this WG's W_hh slice + small constants
    const uint4* src = (const uint4*)(Whh + (size_t)g * 48 * kH);
    uint4* dst = (uint4*)sWhh;
    for (int i = tid; i < 48 * kH / 8; i += 256) dst[i] = src[i];
  }
  for (int i = tid; i < kG; i += 256) { sWf0[i] = wf0g[i]; sBhh[i] = b_hh[i]; }
  for (int i = tid; i < kH; i += 256) sPw[i] = post_w[i];
  const float pb = post_b[0];
  __syncthreads();

  for (int t = 0; t < kT; ++t) {
#ifdef HAVE_TDM
    if (wave == 0) {
      // TDM: async 2D tile load of gi rows b=g*16..g*16+15 for step t into LDS.
      // tile 768 x 16 elems (2B), row stride T*768 elems.
      unsigned long long ga =
          (unsigned long long)(uintptr_t)(gi + ((size_t)(g * 16) * kT + t) * kG);
      unsigned lds = (unsigned)(uintptr_t)(&sGi[0]);
      v4u g0;
      g0[0] = 1u;                                   // count=1 valid descriptor
      g0[1] = lds;                                  // lds_addr
      g0[2] = (unsigned)ga;                         // global_addr[31:0]
      g0[3] = (unsigned)((ga >> 32) & 0x01FFFFFFull) | (2u << 30);  // addr[56:32] | type=2
      v8i g1;
      g1[0] = (int)(1u << 16);                      // data_size=1 (2 bytes)
      g1[1] = (int)(((unsigned)kG & 0xFFFFu) << 16);          // tensor_dim0 lo16
      g1[2] = (int)((16u & 0xFFFFu) << 16);                   // dim0 hi | tensor_dim1 lo=16
      g1[3] = (int)(((unsigned)kG) << 16);                    // dim1 hi | tile_dim0=768
      g1[4] = (int)(16u);                                     // tile_dim1=16, tile_dim2=0
      unsigned long long s0 = (unsigned long long)kT * kG;    // dim0 stride (elems)
      g1[5] = (int)(s0 & 0xFFFFFFFFull);
      g1[6] = (int)((s0 >> 32) & 0xFFFFull);
      g1[7] = 0;
      v4i z4 = {0, 0, 0, 0};
      v8i z8 = {0, 0, 0, 0, 0, 0, 0, 0};
      __builtin_amdgcn_tensor_load_to_lds(g0, g1, z4, z4, z8, 0);
    }
#endif
    // ---- stage 1: gh block for columns [g*48, g*48+48), all 256 batch rows ----
    // wave w owns m-tiles {2w, 2w+1} x all 3 n-tiles: 2 A-frags + 3 B-frags -> 6 wmma/k-step
    {
      int m0 = wave * 32;
      v8f a0b0 = {0.f,0.f,0.f,0.f,0.f,0.f,0.f,0.f};
      v8f a0b1 = a0b0, a0b2 = a0b0, a1b0 = a0b0, a1b1 = a0b0, a1b2 = a0b0;
#pragma unroll
      for (int kk = 0; kk < kH / 32; ++kk) {
        int k0 = kk * 32;
        v16bf a0 = load_a_frag_u128(hidb, kH, m0, k0);
        v16bf a1 = load_a_frag_u128(hidb, kH, m0 + 16, k0);
        v16bf b0 = load_b_frag(sWhh, kH, 0, k0);
        v16bf b1 = load_b_frag(sWhh, kH, 16, k0);
        v16bf b2 = load_b_frag(sWhh, kH, 32, k0);
        a0b0 = wmma_bf16(a0, b0, a0b0);
        a0b1 = wmma_bf16(a0, b1, a0b1);
        a0b2 = wmma_bf16(a0, b2, a0b2);
        a1b0 = wmma_bf16(a1, b0, a1b0);
        a1b1 = wmma_bf16(a1, b1, a1b1);
        a1b2 = wmma_bf16(a1, b2, a1b2);
      }
      v8f acc[2][3] = {{a0b0, a0b1, a0b2}, {a1b0, a1b1, a1b2}};
      int l = tid & 31;
#pragma unroll
      for (int mi = 0; mi < 2; ++mi) {
#pragma unroll
        for (int ni = 0; ni < 3; ++ni) {
          int n = g * 48 + ni * 16 + (l & 15);
          int mb = m0 + mi * 16 + ((l < 16) ? 0 : 8);
#pragma unroll
          for (int r = 0; r < 8; ++r) gh[(size_t)(mb + r) * kG + n] = acc[mi][ni][r];
        }
      }
    }
    grid_barrier(cnt);                 // all gh columns visible everywhere
#ifdef HAVE_TDM
    if (wave == 0) wait_tensorcnt0();  // gi tile landed in LDS
    __syncthreads();
#endif
    // ---- stage 2: gates for rows b = g*16 .. g*16+15 ----
    {
      int rl = tid >> 4;                // local row 0..15
      int b  = g * 16 + rl;
      int h0 = (tid & 15) * 16;         // 16 hidden elems per thread
      float f0b = f0[b];
      const float* ghrow = gh + (size_t)b * kG;
      const float* hrow  = hidf + (size_t)b * kH;
#ifdef HAVE_TDM
      const ushort_t* girow = sGi + (size_t)rl * kG;
#else
      const ushort_t* girow = gi + ((size_t)b * kT + t) * kG;
#endif
      float partial = 0.f;
#pragma unroll
      for (int i = 0; i < 16; ++i) {
        int hh = h0 + i;
        float pre_r = bf2f(girow[hh])        + f0b * sWf0[hh]        + ghrow[hh]        + sBhh[hh];
        float pre_z = bf2f(girow[kH + hh])   + f0b * sWf0[kH + hh]   + ghrow[kH + hh]   + sBhh[kH + hh];
        float r = 1.f / (1.f + __expf(-pre_r));
        float z = 1.f / (1.f + __expf(-pre_z));
        float n = tanhf(bf2f(girow[2 * kH + hh]) + f0b * sWf0[2 * kH + hh] +
                        r * (ghrow[2 * kH + hh] + sBhh[2 * kH + hh]));
        float hn = (1.f - z) * n + z * hrow[hh];
        hidf[(size_t)b * kH + hh] = hn;
        hidb[(size_t)b * kH + hh] = f2bf(hn);
        partial += hn * sPw[hh];
      }
      sRed[tid] = partial;
    }
    __syncthreads();
    if ((tid & 15) == 0) {
      float s = 0.f;
#pragma unroll
      for (int i = 0; i < 16; ++i) s += sRed[tid + i];
      float f0v = s + pb;
      int b = g * 16 + (tid >> 4);
      f0[b] = f0v;
      out[(size_t)b * kT + t] = f0v;
    }
    grid_barrier(cnt);                 // new hidden visible before next matmul
  }
}

// ---------------- host launch ----------------
extern "C" void kernel_launch(void* const* d_in, const int* in_sizes, int n_in,
                              void* d_out, int out_size, void* d_ws, size_t ws_size,
                              hipStream_t stream) {
  (void)in_sizes; (void)n_in; (void)out_size; (void)ws_size;
  const int*   vlist  = (const int*)d_in[1];
  const int*   clist  = (const int*)d_in[2];
  const int*   sa     = (const int*)d_in[3];
  const int*   ea     = (const int*)d_in[4];
  const int*   sap    = (const int*)d_in[5];
  const int*   eap    = (const int*)d_in[6];
  const int*   spk    = (const int*)d_in[7];
  const float* hid0   = (const float*)d_in[8];
  const float* pht    = (const float*)d_in[9];
  const float* spt    = (const float*)d_in[10];
  const float* enc_w  = (const float*)d_in[11];
  const float* enc_b  = (const float*)d_in[12];
  const float* W_ih   = (const float*)d_in[13];
  const float* W_hh   = (const float*)d_in[14];
  const float* b_ih   = (const float*)d_in[15];
  const float* b_hh   = (const float*)d_in[16];
  const float* post_w = (const float*)d_in[17];
  const float* post_b = (const float*)d_in[18];

  char* ws = (char*)d_ws;
  size_t off = 0;
  auto alloc = [&](size_t bytes) -> void* {
    void* p = ws + off;
    off = (off + bytes + 255) & ~(size_t)255;
    return p;
  };
  ushort_t* enc  = (ushort_t*)alloc((size_t)kB * (kT + 2) * kInCh * 2 + 256);
  ushort_t* Bc   = (ushort_t*)alloc((size_t)kC * kKPad * 2);
  ushort_t* Wi   = (ushort_t*)alloc((size_t)kG * kH * 2);
  ushort_t* Whh  = (ushort_t*)alloc((size_t)kG * kH * 2);
  float*    wf0  = (float*)alloc((size_t)kG * 4);
  ushort_t* hbuf = (ushort_t*)alloc((size_t)kM * kC * 2);
  ushort_t* gi   = (ushort_t*)alloc((size_t)kM * kG * 2);
  float*    hidf = (float*)alloc((size_t)kB * kH * 4);
  ushort_t* hidb = (ushort_t*)alloc((size_t)kB * kH * 2);
  float*    gh   = (float*)alloc((size_t)kB * kG * 4);
  float*    f0   = (float*)alloc((size_t)kB * 4);
  unsigned* cnt  = (unsigned*)alloc(256);

  k_prep<<<512, 256, 0, stream>>>(enc_w, W_ih, W_hh, hid0, Bc, Wi, Whh, wf0,
                                  hidf, hidb, f0, cnt);
  k_encin<<<4096, 256, 0, stream>>>(vlist, clist, sa, ea, sap, eap, spk, pht, spt, enc);
  // conv GEMM: 8192 m-blocks x 8 n-blocks (32x32 per wave), 8 waves/block
  k_conv<<<(8192 * 8) / 8, 256, 0, stream>>>(enc, Bc, enc_b, hbuf);
  // gi GEMM: 8192 m-blocks x 24 n-blocks (32x32 per wave)
  k_gi<<<(8192 * 24) / 8, 256, 0, stream>>>(hbuf, Wi, b_ih, gi);
  // sequential GRU: 16 cooperating workgroups
  k_seq<<<kNWG, 256, 0, stream>>>(gi, Whh, wf0, b_hh, post_w, post_b,
                                  hidf, hidb, gh, f0, cnt, (float*)d_out);
}
